// CrystalGNN_57964878627401
// MI455X (gfx1250) — matrix operand order repaired and checked by reading, not compiled, not run on hardware
//
#include <hip/hip_runtime.h>

// CrystalGNN edge-message layer for MI455X (gfx1250, wave32, WMMA).
// 16 edges per wave -> 16x192 H operand -> f16 WMMA GEMMs, f32 accum.
// Layer-1 msg+att fused into one 192x176 GEMM; att 48->24 zero-padded on
// WMMA; att 24->1 via padded-aw3 dot + shfl_xor butterfly landing in the
// C/D row layout so the sigmoid gate applies in registers. B-fragments are
// pre-loaded per output tile (clause the ds_load_b128s, staggered dscnt
// waits across the WMMA chain). The H gather for tile i+1 is issued before
// tile i's atomic epilogue (software pipeline, single A buffer), and the
// once-streamed E matrix uses non-temporal loads to preserve L2 residency
// of X and the atomic-accumulated output.

typedef __attribute__((ext_vector_type(16))) _Float16 v16h;
typedef __attribute__((ext_vector_type(8)))  float    v8f;
typedef __attribute__((ext_vector_type(2)))  float    v2f;

#define DIM   64
#define DIM3  192
#define H1    128           // msg hidden
#define A1    48            // att hidden 1 (padded to 64 in K for layer 2)
#define A2    24            // att hidden 2 (padded to 32 in N)
#define L1N   (H1 + A1)     // fused layer-1 output width = 176
#define WAVES 4
#define TPB   (WAVES * 32)
#define SCR_STRIDE 196      // 176 layer-1 cols + 16 zero-pad + 4 bank stagger

union V16HU {
    v16h      v;
    uint32_t  u[8];
    _Float16  h[16];
};

__global__ void crystal_init_out(const float* __restrict__ X,
                                 float* __restrict__ out, int n4) {
    int i = blockIdx.x * blockDim.x + threadIdx.x;
    if (i < n4) ((float4*)out)[i] = ((const float4*)X)[i];
}

// Gather H = [X[src] | X[dst] | E[e]] row for this lane into six 16x32 f16
// A-fragments. A layout (ISA 7.12.2): m = lane%16; VGPR j<4: k = 8*hi + 2j;
// VGPR j>=4: k = 16 + 8*hi + 2(j-4). k even -> v2f loads, merge to b128.
// E is streamed once -> non-temporal.
__device__ __forceinline__ void gather_H(V16HU Ah[6],
                                         const float* __restrict__ X,
                                         const float* __restrict__ E,
                                         int sidx, int didx, int e, int hi) {
    #pragma unroll
    for (int f = 0; f < 6; ++f) {
        #pragma unroll
        for (int j = 0; j < 8; ++j) {
            int k = f * 32 + (j < 4 ? hi * 8 + 2 * j : 16 + hi * 8 + 2 * (j - 4));
            v2f xv;
            if (k < DIM)          xv = *(const v2f*)(X + (size_t)sidx * DIM + k);
            else if (k < 2 * DIM) xv = *(const v2f*)(X + (size_t)didx * DIM + (k - DIM));
            else                  xv = __builtin_nontemporal_load(
                                          (const v2f*)(E + (size_t)e * DIM + (k - 2 * DIM)));
            Ah[f].h[2 * j]     = (_Float16)xv.x;
            Ah[f].h[2 * j + 1] = (_Float16)xv.y;
        }
    }
}

__global__ __launch_bounds__(TPB)
void crystal_gnn_edge(const float* __restrict__ X, const float* __restrict__ E,
                      const int* __restrict__ ei, int n_edges,
                      const float* __restrict__ aw1, const float* __restrict__ ab1,
                      const float* __restrict__ aw2, const float* __restrict__ ab2,
                      const float* __restrict__ aw3, const float* __restrict__ ab3,
                      const float* __restrict__ mw1, const float* __restrict__ mb1,
                      const float* __restrict__ mw2, const float* __restrict__ mb2,
                      float* __restrict__ out)
{
    // Weights transposed (Wt[n][k]) in f16 so a B-fragment is two b128 LDS loads.
    __shared__ __align__(16) _Float16 s_w1t[L1N * DIM3];   // [mw1 | aw1] fused
    __shared__ __align__(16) _Float16 s_mw2t[DIM * H1];
    __shared__ __align__(16) _Float16 s_aw2t[32 * 64];     // padded: N 24->32, K 48->64
    __shared__ float s_aw3p[32];                           // zero-padded
    __shared__ float s_b1[L1N], s_mb2[DIM], s_ab2p[32], s_ab3;
    __shared__ float s_scr[WAVES][16 * SCR_STRIDE];        // per-wave layer-1 outputs
    __shared__ int   s_dst[WAVES][16];

    const int tid = threadIdx.x;
    for (int i = tid; i < H1 * DIM3; i += TPB) { int k = i / H1, n = i % H1; s_w1t[n * DIM3 + k] = (_Float16)mw1[i]; }
    for (int i = tid; i < A1 * DIM3; i += TPB) { int k = i / A1, n = i % A1; s_w1t[(H1 + n) * DIM3 + k] = (_Float16)aw1[i]; }
    for (int i = tid; i < DIM * H1; i += TPB)  { int k = i / DIM, n = i % DIM; s_mw2t[n * H1 + k] = (_Float16)mw2[i]; }
    for (int i = tid; i < 32 * 64; i += TPB) {
        int n = i >> 6, k = i & 63;
        s_aw2t[i] = (n < A2 && k < A1) ? (_Float16)aw2[k * A2 + n] : (_Float16)0.0f;
    }
    for (int i = tid; i < 32; i += TPB) {
        s_aw3p[i] = (i < A2) ? aw3[i] : 0.0f;
        s_ab2p[i] = (i < A2) ? ab2[i] : 0.0f;
    }
    for (int i = tid; i < H1; i += TPB) s_b1[i] = mb1[i];
    for (int i = tid; i < A1; i += TPB) s_b1[H1 + i] = ab1[i];
    for (int i = tid; i < DIM; i += TPB) s_mb2[i] = mb2[i];
    if (tid == 0) s_ab3 = ab3[0];
    __syncthreads();

    const int wave = tid >> 5;
    const int lane = tid & 31;
    const int n16  = lane & 15;   // N / row-within-tile index
    const int hi   = lane >> 4;   // half-wave select in WMMA layouts
    const int ntiles = n_edges >> 4;
    float* scr = s_scr[wave];

    // Zero the 16 K-pad columns (176..191) once; tile loop never writes them.
    for (int i = lane; i < 16 * 16; i += 32)
        scr[(i >> 4) * SCR_STRIDE + 176 + (i & 15)] = 0.0f;

    // Hoist loop-invariant per-lane bias/weight values.
    float b1v[11], mb2v[4], ab2v[2];
    #pragma unroll
    for (int i = 0; i < 11; ++i) b1v[i] = s_b1[i * 16 + n16];
    #pragma unroll
    for (int i = 0; i < 4; ++i) mb2v[i] = s_mb2[i * 16 + n16];
    #pragma unroll
    for (int i = 0; i < 2; ++i) ab2v[i] = s_ab2p[i * 16 + n16];
    const float w3a  = s_aw3p[n16];
    const float w3b  = s_aw3p[16 + n16];
    const float ab3v = s_ab3;

    const int tstride = gridDim.x * WAVES;
    int t = blockIdx.x * WAVES + wave;
    if (t < ntiles) {
        // ---- pipeline prologue: gather first tile's H
        int e    = t * 16 + n16;
        int sidx = ei[e];
        int didx = ei[n_edges + e];
        V16HU Ah[6];
        gather_H(Ah, X, E, sidx, didx, e, hi);

        while (t < ntiles) {
            if (lane < 16) s_dst[wave][n16] = didx;

            // Next tile's indices (clamped on the final iteration so the body
            // stays one straight-line block; the extra gather is harmless).
            const int tn = t + tstride;
            const int tc = (tn < ntiles) ? tn : t;
            const int eN = tc * 16 + n16;
            const int sN = ei[eN];
            const int dN = ei[n_edges + eN];

            // ---- Fused layer 1: relu(H @ [mw1|aw1] + [mb1|ab1]) -> scratch [0,176)
            #pragma unroll
            for (int nt = 0; nt < L1N / 16; ++nt) {
                V16HU Bfs[6];
                #pragma unroll
                for (int f = 0; f < 6; ++f)
                    #pragma unroll
                    for (int j = 0; j < 8; ++j)
                        Bfs[f].u[j] = *(const uint32_t*)&s_w1t[(nt * 16 + n16) * DIM3 + f * 32 + (j < 4 ? hi * 8 + 2 * j : 16 + hi * 8 + 2 * (j - 4))];
                v8f acc;
                #pragma unroll
                for (int r = 0; r < 8; ++r) acc[r] = b1v[nt];
                #pragma unroll
                for (int f = 0; f < 6; ++f)
                    acc = __builtin_amdgcn_wmma_f32_16x16x32_f16(false, Ah[f].v, false, Bfs[f].v,
                                                                 (short)0, acc, false, false);
                #pragma unroll
                for (int r = 0; r < 8; ++r)
                    scr[(r + 8 * hi) * SCR_STRIDE + nt * 16 + n16] = fmaxf(acc[r], 0.0f);
            }

            // ---- re-layout h1 (16x128) from scratch into four A-fragments
            V16HU Am[4];
            #pragma unroll
            for (int f = 0; f < 4; ++f) {
                #pragma unroll
                for (int j = 0; j < 8; ++j) {
                    int k = f * 32 + (j < 4 ? hi * 8 + 2 * j : 16 + hi * 8 + 2 * (j - 4));
                    Am[f].h[2 * j]     = (_Float16)scr[n16 * SCR_STRIDE + k];
                    Am[f].h[2 * j + 1] = (_Float16)scr[n16 * SCR_STRIDE + k + 1];
                }
            }

            // ---- msg layer 2: h1 @ mw2 + mb2  -> macc (16x64)
            v8f macc[4];
            #pragma unroll
            for (int nt = 0; nt < 4; ++nt) {
                V16HU Bfs[4];
                #pragma unroll
                for (int f = 0; f < 4; ++f)
                    #pragma unroll
                    for (int j = 0; j < 8; ++j)
                        Bfs[f].u[j] = *(const uint32_t*)&s_mw2t[(nt * 16 + n16) * H1 + f * 32 + (j < 4 ? hi * 8 + 2 * j : 16 + hi * 8 + 2 * (j - 4))];
                #pragma unroll
                for (int r = 0; r < 8; ++r) macc[nt][r] = mb2v[nt];
                #pragma unroll
                for (int f = 0; f < 4; ++f)
                    macc[nt] = __builtin_amdgcn_wmma_f32_16x16x32_f16(false, Am[f].v, false, Bfs[f].v,
                                                                      (short)0, macc[nt], false, false);
            }

            // ---- att layer 2 on WMMA: a1_pad @ aw2_pad + ab2
            // (A: scratch cols [128,192) = 48 real + 16 zero-pad K columns)
            V16HU Aa[2];
            #pragma unroll
            for (int f = 0; f < 2; ++f) {
                #pragma unroll
                for (int j = 0; j < 8; ++j) {
                    int k = f * 32 + (j < 4 ? hi * 8 + 2 * j : 16 + hi * 8 + 2 * (j - 4));
                    Aa[f].h[2 * j]     = (_Float16)scr[n16 * SCR_STRIDE + 128 + k];
                    Aa[f].h[2 * j + 1] = (_Float16)scr[n16 * SCR_STRIDE + 128 + k + 1];
                }
            }
            v8f acc2[2];
            #pragma unroll
            for (int nt = 0; nt < 2; ++nt) {
                V16HU Bfs[2];
                #pragma unroll
                for (int f = 0; f < 2; ++f)
                    #pragma unroll
                    for (int j = 0; j < 8; ++j)
                        Bfs[f].u[j] = *(const uint32_t*)&s_aw2t[(nt * 16 + n16) * 64 + f * 32 + (j < 4 ? hi * 8 + 2 * j : 16 + hi * 8 + 2 * (j - 4))];
                #pragma unroll
                for (int r = 0; r < 8; ++r) acc2[nt][r] = ab2v[nt];
                #pragma unroll
                for (int f = 0; f < 2; ++f)
                    acc2[nt] = __builtin_amdgcn_wmma_f32_16x16x32_f16(false, Aa[f].v, false, Bfs[f].v,
                                                                      (short)0, acc2[nt], false, false);
            }

            // ---- software pipeline: issue next tile's H gather now; the loads
            // complete under the shuffle/sigmoid/atomic epilogue below.
            gather_H(Ah, X, E, sN, dN, eN, hi);

            // ---- att layer 3 (24->1, aw3 zero-padded) + sigmoid gate + scatter.
            // shfl_xor butterfly over the 16-lane half sums row m = r + 8*hi --
            // the same row macc[*][r] holds, so the gate applies in registers.
            #pragma unroll
            for (int r = 0; r < 8; ++r) {
                float p = fmaxf(acc2[0][r], 0.0f) * w3a + fmaxf(acc2[1][r], 0.0f) * w3b;
                p += __shfl_xor(p, 1, 32);
                p += __shfl_xor(p, 2, 32);
                p += __shfl_xor(p, 4, 32);
                p += __shfl_xor(p, 8, 32);
                const float sig = 1.0f / (1.0f + __expf(-(p + ab3v)));
                const int d = s_dst[wave][r + 8 * hi];
                #pragma unroll
                for (int nt = 0; nt < 4; ++nt)
                    unsafeAtomicAdd(&out[(size_t)d * DIM + nt * 16 + n16], macc[nt][r] * sig);
            }

            sidx = sN; didx = dN; e = eN; t = tn;
        }
    }
}

extern "C" void kernel_launch(void* const* d_in, const int* in_sizes, int n_in,
                              void* d_out, int out_size, void* d_ws, size_t ws_size,
                              hipStream_t stream) {
    const float* X   = (const float*)d_in[0];
    const float* E   = (const float*)d_in[1];
    // d_in[2] emb_nodes, d_in[3] emb_edges: unused by the reference math
    const int*   ei  = (const int*)d_in[4];
    const float* aw1 = (const float*)d_in[5];
    const float* ab1 = (const float*)d_in[6];
    const float* aw2 = (const float*)d_in[7];
    const float* ab2 = (const float*)d_in[8];
    const float* aw3 = (const float*)d_in[9];
    const float* ab3 = (const float*)d_in[10];
    const float* mw1 = (const float*)d_in[11];
    const float* mb1 = (const float*)d_in[12];
    const float* mw2 = (const float*)d_in[13];
    const float* mb2 = (const float*)d_in[14];
    float* out = (float*)d_out;

    const int n_nodes = in_sizes[0] / DIM;
    const int n_edges = in_sizes[1] / DIM;

    const int n4 = (n_nodes * DIM) / 4;
    crystal_init_out<<<(n4 + 255) / 256, 256, 0, stream>>>(X, out, n4);

    crystal_gnn_edge<<<1024, TPB, 0, stream>>>(X, E, ei, n_edges,
                                               aw1, ab1, aw2, ab2, aw3, ab3,
                                               mw1, mb1, mw2, mb2, out);
}
